// Attention_2645699854594
// MI455X (gfx1250) — matrix operand order repaired and checked by reading, not compiled, not run on hardware
//
#include <hip/hip_runtime.h>
#include <hip/hip_bf16.h>
#include <stdint.h>

typedef __bf16 v16bf  __attribute__((ext_vector_type(16)));
typedef __bf16 bf16x8 __attribute__((ext_vector_type(8)));
typedef float  v8f    __attribute__((ext_vector_type(8)));
typedef float  f32x8  __attribute__((ext_vector_type(8)));

constexpr int BATCH = 4;
constexpr int SEQ   = 2048;
constexpr int DIM   = 1024;
constexpr int DQKV  = 1024;

#define WMMA_BF16(A, B, C) \
    __builtin_amdgcn_wmma_f32_16x16x32_bf16(false, (A), false, (B), (short)0, (C), false, false)

// Per-lane 16B global -> LDS async copy (CDNA5 GLOBAL_LOAD_ASYNC_TO_LDS_B128).
// vdst = VGPR with LDS byte address (flat-address low 32 bits == LDS offset, ISA 10.2).
__device__ __forceinline__ void async_copy_b128(const void* gsrc, void* lds_dst) {
    unsigned lds = (unsigned)(uintptr_t)lds_dst;
    asm volatile("global_load_async_to_lds_b128 %0, %1, off"
                 :: "v"(lds), "v"(gsrc)
                 : "memory");
}
__device__ __forceinline__ void wait_async_zero() {
    asm volatile("s_wait_asynccnt 0x0" ::: "memory");
}

// Build a 16-bit-A fragment (16x32, lane l: row l%16, K-chunks [8*(l/16),+8) and [16+8*(l/16),+8))
__device__ __forceinline__ v16bf make_afrag(bf16x8 lo, bf16x8 hi) {
    v16bf a;
#pragma unroll
    for (int j = 0; j < 8; ++j) { a[j] = lo[j]; a[j + 8] = hi[j]; }
    return a;
}

// ---------------------------------------------------------------------------
// Kernel 1: QKV projection.  C = X(f32) @ W(f32) -> bf16.
//   MODE: 0 -> Q (scaled by 1/32), 1 -> K (stored transposed), 2 -> V
//   block = 256 threads (8 waves); wave tile 32x64; block tile 256x64; K-step 32.
//   (B tile staged through VGPRs because of the f32 -> bf16 conversion.)
// ---------------------------------------------------------------------------
template <int MODE>
__global__ __launch_bounds__(256)
void qkv_gemm(const float* __restrict__ x,
              const float* __restrict__ w,
              __bf16* __restrict__ dst)
{
    __shared__ __align__(16) __bf16 Bt[2][32 * 64];

    const int row0 = blockIdx.x * 256;
    const int col0 = blockIdx.y * 64;
    const int tid  = threadIdx.x;
    const int wave = tid >> 5;
    const int lane = tid & 31;
    const int half = lane >> 4;
    const int l16  = lane & 15;
    const int mrow0 = row0 + wave * 32 + l16;
    const int mrow1 = mrow0 + 16;

    const int kr = tid >> 3;
    const int c8 = (tid & 7) * 8;

    auto stage = [&](int k0, int buf) {
        f32x8 wv8 = *(const f32x8*)(w + (size_t)(k0 + kr) * DQKV + col0 + c8);
        *(bf16x8*)(&Bt[buf][kr * 64 + c8]) = __builtin_convertvector(wv8, bf16x8);
    };

    v8f acc[2][4];
#pragma unroll
    for (int r = 0; r < 2; ++r)
#pragma unroll
        for (int nt = 0; nt < 4; ++nt) acc[r][nt] = {};

    stage(0, 0);
    int cur = 0;
    for (int k0 = 0; k0 < DIM; k0 += 32) {
        __syncthreads();
        if (k0 + 32 < DIM) stage(k0 + 32, cur ^ 1);

        const float* xp0 = x + (size_t)mrow0 * DIM + k0 + 8 * half;
        const float* xp1 = x + (size_t)mrow1 * DIM + k0 + 8 * half;
        v16bf a0 = make_afrag(__builtin_convertvector(*(const f32x8*)(xp0),      bf16x8),
                              __builtin_convertvector(*(const f32x8*)(xp0 + 16), bf16x8));
        v16bf a1 = make_afrag(__builtin_convertvector(*(const f32x8*)(xp1),      bf16x8),
                              __builtin_convertvector(*(const f32x8*)(xp1 + 16), bf16x8));

        const __bf16* bp = &Bt[cur][lane * 64];
#pragma unroll
        for (int nt = 0; nt < 4; ++nt) {
            v16bf bfrag;
#pragma unroll
            for (int s = 0; s < 16; ++s) bfrag[s] = Bt[cur][(8 * half + (s & 7) + ((s >> 3) << 4)) * 64 + nt * 16 + l16];
            acc[0][nt] = WMMA_BF16(a0, bfrag, acc[0][nt]);
            acc[1][nt] = WMMA_BF16(a1, bfrag, acc[1][nt]);
        }
        cur ^= 1;
    }

    const float scale = (MODE == 0) ? 0.03125f : 1.0f;   // fold 1/sqrt(1024) into Q
#pragma unroll
    for (int r = 0; r < 2; ++r)
#pragma unroll
        for (int nt = 0; nt < 4; ++nt)
#pragma unroll
            for (int i = 0; i < 8; ++i) {
                float val = acc[r][nt][i] * scale;
                int m = row0 + wave * 32 + r * 16 + i + 8 * half;
                int n = col0 + nt * 16 + l16;
                if constexpr (MODE == 1) {       // K: store transposed Kt[b][n][s]
                    int b = m >> 11, s = m & (SEQ - 1);
                    dst[((size_t)b * DQKV + n) * SEQ + s] = (__bf16)val;
                } else {
                    dst[(size_t)m * DQKV + n] = (__bf16)val;
                }
            }
}

// ---------------------------------------------------------------------------
// Kernel 2: scores[b] = Q[b] @ Kt[b]  -> bf16 (scale folded into Q already)
//   B tile staged with async global->LDS copies (ASYNCcnt).
// ---------------------------------------------------------------------------
__global__ __launch_bounds__(256)
void scores_gemm(const __bf16* __restrict__ Q,
                 const __bf16* __restrict__ Kt,
                 __bf16* __restrict__ Sc)
{
    __shared__ __align__(16) __bf16 Bt[2][32 * 64];

    const int b    = blockIdx.z;
    const int row0 = blockIdx.x * 256;
    const int col0 = blockIdx.y * 64;
    const int tid  = threadIdx.x;
    const int wave = tid >> 5;
    const int lane = tid & 31;
    const int half = lane >> 4;
    const int l16  = lane & 15;
    const int mrow0 = row0 + wave * 32 + l16;
    const int mrow1 = mrow0 + 16;

    const int kr = tid >> 3;
    const int c8 = (tid & 7) * 8;

    auto stage = [&](int k0, int buf) {
        async_copy_b128(Kt + ((size_t)b * DQKV + k0 + kr) * SEQ + col0 + c8,
                        &Bt[buf][kr * 64 + c8]);
    };

    v8f acc[2][4];
#pragma unroll
    for (int r = 0; r < 2; ++r)
#pragma unroll
        for (int nt = 0; nt < 4; ++nt) acc[r][nt] = {};

    stage(0, 0);
    int cur = 0;
    for (int k0 = 0; k0 < DQKV; k0 += 32) {
        wait_async_zero();          // my previous stage landed in LDS
        __syncthreads();            // everyone's did
        if (k0 + 32 < DQKV) stage(k0 + 32, cur ^ 1);

        const __bf16* qp0 = Q + ((size_t)(b * SEQ) + mrow0) * DQKV + k0 + 8 * half;
        const __bf16* qp1 = Q + ((size_t)(b * SEQ) + mrow1) * DQKV + k0 + 8 * half;
        v16bf a0 = make_afrag(*(const bf16x8*)(qp0), *(const bf16x8*)(qp0 + 16));
        v16bf a1 = make_afrag(*(const bf16x8*)(qp1), *(const bf16x8*)(qp1 + 16));

        const __bf16* bp = &Bt[cur][lane * 64];
#pragma unroll
        for (int nt = 0; nt < 4; ++nt) {
            v16bf bfrag;
#pragma unroll
            for (int s = 0; s < 16; ++s) bfrag[s] = Bt[cur][(8 * half + (s & 7) + ((s >> 3) << 4)) * 64 + nt * 16 + l16];
            acc[0][nt] = WMMA_BF16(a0, bfrag, acc[0][nt]);
            acc[1][nt] = WMMA_BF16(a1, bfrag, acc[1][nt]);
        }
        cur ^= 1;
    }

#pragma unroll
    for (int r = 0; r < 2; ++r)
#pragma unroll
        for (int nt = 0; nt < 4; ++nt)
#pragma unroll
            for (int i = 0; i < 8; ++i) {
                int m = row0 + wave * 32 + r * 16 + i + 8 * half;
                int n = col0 + nt * 16 + l16;
                Sc[((size_t)(b * SEQ) + m) * SEQ + n] = (__bf16)acc[r][nt][i];
            }
}

// ---------------------------------------------------------------------------
// Kernel 3: in-place row softmax over Sc (one block per row, 2048 cols)
// ---------------------------------------------------------------------------
__global__ __launch_bounds__(256)
void softmax_rows(__bf16* __restrict__ Sc)
{
    __shared__ float red[256];
    const int tid = threadIdx.x;
    __bf16* rp = Sc + (size_t)blockIdx.x * SEQ;

    bf16x8 v = *(const bf16x8*)(rp + tid * 8);
    f32x8  f = __builtin_convertvector(v, f32x8);

    float m = f[0];
#pragma unroll
    for (int j = 1; j < 8; ++j) m = fmaxf(m, f[j]);
    red[tid] = m;
    __syncthreads();
    for (int s = 128; s > 0; s >>= 1) {
        if (tid < s) red[tid] = fmaxf(red[tid], red[tid + s]);
        __syncthreads();
    }
    m = red[0];
    __syncthreads();

    f32x8 e;
    float sum = 0.f;
#pragma unroll
    for (int j = 0; j < 8; ++j) { e[j] = __expf(f[j] - m); sum += e[j]; }
    red[tid] = sum;
    __syncthreads();
    for (int s = 128; s > 0; s >>= 1) {
        if (tid < s) red[tid] += red[tid + s];
        __syncthreads();
    }
    float inv = 1.0f / red[0];
#pragma unroll
    for (int j = 0; j < 8; ++j) e[j] *= inv;
    *(bf16x8*)(rp + tid * 8) = __builtin_convertvector(e, bf16x8);
}

// ---------------------------------------------------------------------------
// Kernel 4: out[b] = P[b] @ V[b]  -> f32 d_out   (K-dim = SEQ = 2048)
//   B tile staged with async global->LDS copies (ASYNCcnt).
// ---------------------------------------------------------------------------
__global__ __launch_bounds__(256)
void out_gemm(const __bf16* __restrict__ P,
              const __bf16* __restrict__ V,
              float* __restrict__ out)
{
    __shared__ __align__(16) __bf16 Bt[2][32 * 64];

    const int b    = blockIdx.z;
    const int row0 = blockIdx.x * 256;
    const int col0 = blockIdx.y * 64;
    const int tid  = threadIdx.x;
    const int wave = tid >> 5;
    const int lane = tid & 31;
    const int half = lane >> 4;
    const int l16  = lane & 15;
    const int mrow0 = row0 + wave * 32 + l16;
    const int mrow1 = mrow0 + 16;

    const int kr = tid >> 3;
    const int c8 = (tid & 7) * 8;

    auto stage = [&](int k0, int buf) {
        async_copy_b128(V + ((size_t)(b * SEQ) + k0 + kr) * DQKV + col0 + c8,
                        &Bt[buf][kr * 64 + c8]);
    };

    v8f acc[2][4];
#pragma unroll
    for (int r = 0; r < 2; ++r)
#pragma unroll
        for (int nt = 0; nt < 4; ++nt) acc[r][nt] = {};

    stage(0, 0);
    int cur = 0;
    for (int k0 = 0; k0 < SEQ; k0 += 32) {
        wait_async_zero();
        __syncthreads();
        if (k0 + 32 < SEQ) stage(k0 + 32, cur ^ 1);

        const __bf16* pp0 = P + ((size_t)(b * SEQ) + mrow0) * SEQ + k0 + 8 * half;
        const __bf16* pp1 = P + ((size_t)(b * SEQ) + mrow1) * SEQ + k0 + 8 * half;
        v16bf a0 = make_afrag(*(const bf16x8*)(pp0), *(const bf16x8*)(pp0 + 16));
        v16bf a1 = make_afrag(*(const bf16x8*)(pp1), *(const bf16x8*)(pp1 + 16));

        const __bf16* bp = &Bt[cur][lane * 64];
#pragma unroll
        for (int nt = 0; nt < 4; ++nt) {
            v16bf bfrag;
#pragma unroll
            for (int s = 0; s < 16; ++s) bfrag[s] = Bt[cur][(8 * half + (s & 7) + ((s >> 3) << 4)) * 64 + nt * 16 + l16];
            acc[0][nt] = WMMA_BF16(a0, bfrag, acc[0][nt]);
            acc[1][nt] = WMMA_BF16(a1, bfrag, acc[1][nt]);
        }
        cur ^= 1;
    }

#pragma unroll
    for (int r = 0; r < 2; ++r)
#pragma unroll
        for (int nt = 0; nt < 4; ++nt)
#pragma unroll
            for (int i = 0; i < 8; ++i) {
                int m = row0 + wave * 32 + r * 16 + i + 8 * half;
                int n = col0 + nt * 16 + l16;
                out[((size_t)(b * SEQ) + m) * DQKV + n] = acc[r][nt][i];
            }
}

// ---------------------------------------------------------------------------
extern "C" void kernel_launch(void* const* d_in, const int* in_sizes, int n_in,
                              void* d_out, int out_size, void* d_ws, size_t ws_size,
                              hipStream_t stream)
{
    (void)in_sizes; (void)n_in; (void)out_size; (void)ws_size;

    const float* x  = (const float*)d_in[0];
    const float* wq = (const float*)d_in[1];
    const float* wk = (const float*)d_in[2];
    const float* wv = (const float*)d_in[3];

    char* ws = (char*)d_ws;
    constexpr size_t QKV_BYTES = (size_t)BATCH * SEQ * DQKV * sizeof(__bf16); // 16.8 MB
    __bf16* Q  = (__bf16*)(ws);
    __bf16* Kt = (__bf16*)(ws + QKV_BYTES);
    __bf16* V  = (__bf16*)(ws + 2 * QKV_BYTES);
    __bf16* Sc = (__bf16*)(ws + 3 * QKV_BYTES);                               // 33.5 MB

    float* out = (float*)d_out;

    dim3 blk(256);
    dim3 gQKV((BATCH * SEQ) / 256, DQKV / 64, 1);
    qkv_gemm<0><<<gQKV, blk, 0, stream>>>(x, wq, Q);
    qkv_gemm<1><<<gQKV, blk, 0, stream>>>(x, wk, Kt);
    qkv_gemm<2><<<gQKV, blk, 0, stream>>>(x, wv, V);

    scores_gemm <<<dim3(SEQ / 256, SEQ / 64,  BATCH), blk, 0, stream>>>(Q, Kt, Sc);
    softmax_rows<<<dim3(BATCH * SEQ),                 blk, 0, stream>>>(Sc);
    out_gemm    <<<dim3(SEQ / 256, DQKV / 64, BATCH), blk, 0, stream>>>(Sc, V, out);
}
